// PS_APM_Seq_46033459479293
// MI455X (gfx1250) — matrix operand order repaired and checked by reading, not compiled, not run on hardware
//
// MI455X (gfx1250, CDNA5) implementation — wave32 + WMMA f32_16x16x32_f16.
//
// Roofline: ~70 GFLOP total vs ~300 MB of f16 intermediates => compute-bound
// GEMMs -> WMMA path. Weights (<=2 MB) are L2-resident; per-step B-fragments
// read from global. Serial GRU/decoder state kept in LDS (f16 mirror for
// WMMA A-frags, f32 master for carry).
//
// Round-3 change:
//  * wmma_acc K-chunk loop unroll 1 -> 2: round-2 disasm showed a clean
//    no-spill loop but with s_wait_loadcnt_dscnt 0x0 immediately before each
//    WMMA (zero load/compute overlap). Unroll 2 double-buffers fragments
//    (<=32 extra VGPRs live) so chunk k+1 loads issue under chunk k's WMMA,
//    without recreating round-1's full-unroll scratch spills.
//
// Workspace layout (bytes):
//   A        : 64*16*16 f32                        @ 0
//   Wn16     : 128*160 f16 (spectral-normalized)   @ 65536
//   Wself16  : 128*160 f16                         @ 106496
//   Wih16    : 16*384*32 f16 (K padded 16->32)     @ 147456
//   Whh16    : 16*384*128 f16                      @ 540672
//   Wk16/Wv16: 128*128 f16                         @ 2113536 / 2146304
//   cWih16   : 384*160 f16 (K padded 144->160)     @ 2179072
//   cWhh16   : 384*128 f16                         @ 2301952
//   readW16  : 16*128 f16                          @ 2400256
//   h0       : 1024*128 f32                        @ 2404352
//   Hc       : 64*16*256*160 f16                   @ 2928640  (84 MB)
//   K        : 64*16*256*128 f16                   @ 86814720 (67 MB)
//   V        : 64*16*256*128 f16                   @ 153923584 (67 MB)

#include <hip/hip_runtime.h>
#include <hip/hip_bf16.h>
#include <stdint.h>

typedef __attribute__((ext_vector_type(16))) _Float16 v16h;
typedef __attribute__((ext_vector_type(8)))  float    v8f;

#define DEV __device__ __forceinline__

// Async global->LDS staging (CDNA5 ASYNCcnt path). LDS byte address = low 32
// bits of the generic pointer (ISA 10.2 aperture truncation).
#define USE_ASYNC_LDS 1

DEV void async_ld_b64(void* lds, const void* gsrc) {
#if USE_ASYNC_LDS
  unsigned l = (unsigned)(uintptr_t)lds;
  asm volatile("global_load_async_to_lds_b64 %0, %1, off"
               :: "v"(l), "v"(gsrc) : "memory");
#else
  *(unsigned long long*)lds = *(const unsigned long long*)gsrc;
#endif
}

DEV void async_wait0() {
#if USE_ASYNC_LDS
  asm volatile("s_wait_asynccnt 0" ::: "memory");
#endif
}

// ---- WMMA fragment helpers (layouts per CDNA5 ISA 7.12.2, wave32) ----------

// A-matrix 16x32 f16 chunk from row-major [16 x ld] source.
// lane L: m = L%16, half = L/16; pair p: kbase = p<4 ? 2p+8*half : 16+2(p-4)+8*half
DEV v16h load_A_frag(const _Float16* base, int ld) {
  int lane = threadIdx.x & 31;
  int m = lane & 15;
  int half = lane >> 4;
  const _Float16* row = base + m * ld;
  v16h a;
#pragma unroll
  for (int p = 0; p < 8; ++p) {
    int kb = (p < 4) ? (2 * p + 8 * half) : (16 + 2 * (p - 4) + 8 * half);
    a[2 * p]     = row[kb];
    a[2 * p + 1] = row[kb + 1];
  }
  return a;
}

// B-matrix 32x16 f16 chunk where B[k][n] = W[n*ld + k] (we compute X @ W^T).
// lane L: n = L%16, half = L/16; pair p: kbase = p<4 ? 2p+8*half : 16+2(p-4)+8*half
DEV v16h load_B_frag(const _Float16* base, int ld) {
  int lane = threadIdx.x & 31;
  int n = lane & 15;
  int half = lane >> 4;
  const _Float16* col = base + n * ld + 8 * half;
  v16h b;
#pragma unroll
  for (int p = 0; p < 8; ++p) {
    int kb = (p < 4) ? (2 * p) : (16 + 2 * (p - 4));
    b[2 * p]     = col[kb];
    b[2 * p + 1] = col[kb + 1];
  }
  return b;
}

// D/C: lane L: n = L%16, half = L/16; element r -> (m = r + 8*half, n)
// unroll 2 = double-buffered fragments: chunk k+1 loads overlap chunk k WMMA
// without the full-unroll register blowup that spilled in round 1.
DEV v8f wmma_acc(const _Float16* Abase, int lda,
                 const _Float16* Bbase, int ldb,
                 int kchunks, v8f acc) {
#pragma unroll 2
  for (int kc = 0; kc < kchunks; ++kc) {
    v16h a = load_A_frag(Abase + kc * 32, lda);
    v16h b = load_B_frag(Bbase + kc * 32, ldb);
    acc = __builtin_amdgcn_wmma_f32_16x16x32_f16(
        /*neg_a=*/false, a, /*neg_b=*/false, b,
        /*c_mod=*/(short)0, acc, /*reuse_a=*/false, /*reuse_b=*/false);
  }
  return acc;
}

DEV float dsigmoid(float x) { return 1.f / (1.f + __expf(-x)); }

// ---- prep kernels ----------------------------------------------------------

// Per-batch phase graph A[b,i,j] from S, G, phases.
__global__ void k_graphs(const float* __restrict S, const float* __restrict G,
                         const int* __restrict phases, float* __restrict A) {
  int b = blockIdx.x;
  int p = phases[b];
  __shared__ float rs[16];
  __shared__ float gv[16];
  __shared__ float gs;
  int tid = threadIdx.x;
  int i = tid >> 4, j = tid & 15;
  if (tid < 16) rs[tid] = 0.f;
  if (tid == 0) gs = 0.f;
  __syncthreads();
  float sz = (i == j) ? 0.f : S[(p * 16 + i) * 16 + j];
  atomicAdd(&rs[i], fabsf(sz));
  if (tid < 16) {
    float g = log1pf(__expf(G[p * 16 + tid])) + 1e-6f;
    gv[tid] = g;
    atomicAdd(&gs, g);
  }
  __syncthreads();
  float denom = fmaxf(rs[i], 1e-6f);
  float gn = gv[i] * (16.f / fmaxf(gs, 1e-6f));
  A[(b * 16 + i) * 16 + j] = (sz / denom) * gn;
}

// Spectral norm of W_neigh (128x136), 15 power iterations; write f16 padded 160.
__global__ void k_spectral(const float* __restrict W, _Float16* __restrict Wn16) {
  __shared__ float u[128];
  __shared__ float v[136];
  __shared__ float acc;
  int tid = threadIdx.x;  // 256
  if (tid < 128) u[tid] = 1.f / sqrtf(128.f);
  __syncthreads();
  for (int it = 0; it < 15; ++it) {
    if (tid == 0) acc = 0.f;
    __syncthreads();
    if (tid < 136) {
      float s = 0.f;
      for (int i = 0; i < 128; ++i) s += W[i * 136 + tid] * u[i];
      v[tid] = s;
      atomicAdd(&acc, s * s);
    }
    __syncthreads();
    float nv = sqrtf(acc);
    if (tid < 136) v[tid] /= nv;
    __syncthreads();
    if (tid == 0) acc = 0.f;
    __syncthreads();
    if (tid < 128) {
      float s = 0.f;
      for (int j = 0; j < 136; ++j) s += W[tid * 136 + j] * v[j];
      u[tid] = s;
      atomicAdd(&acc, s * s);
    }
    __syncthreads();
    float nu = sqrtf(acc);
    if (tid < 128) u[tid] /= nu;
    __syncthreads();
  }
  if (tid == 0) acc = 0.f;
  __syncthreads();
  if (tid < 128) {
    float s = 0.f;
    for (int j = 0; j < 136; ++j) s += W[tid * 136 + j] * v[j];
    atomicAdd(&acc, u[tid] * s);
  }
  __syncthreads();
  float sigma = acc;
  for (int idx = tid; idx < 128 * 160; idx += 256) {
    int r = idx / 160, c = idx % 160;
    Wn16[idx] = (c < 136) ? (_Float16)(W[r * 136 + c] / sigma) : (_Float16)0.f;
  }
}

// Generic f32 -> f16 row-pad convert: [R x Cs] -> [R x Cd] (zero pad).
__global__ void k_padcvt(const float* __restrict src, _Float16* __restrict dst,
                         int R, int Cs, int Cd) {
  int total = R * Cd;
  for (int i = blockIdx.x * blockDim.x + threadIdx.x; i < total;
       i += gridDim.x * blockDim.x) {
    int r = i / Cd, c = i % Cd;
    dst[i] = (c < Cs) ? (_Float16)src[r * Cs + c] : (_Float16)0.f;
  }
}

// ---- GRU over T=256 (one block per region n, 512 threads = 16 waves) -------
__global__ __launch_bounds__(512) void k_gru(
    const float* __restrict X, const _Float16* __restrict Wih16,
    const _Float16* __restrict Whh16, const float* __restrict bih_all,
    const float* __restrict bhh_all, const float* __restrict lng,
    const float* __restrict lnb, const float* __restrict tW,
    const float* __restrict tb, _Float16* __restrict Hc) {
  int n = blockIdx.x;
  int tid = threadIdx.x;
  int w = tid >> 5;
  __shared__ _Float16 sWih[384 * 32];  // 24 KB
  __shared__ _Float16 sXt[64 * 32];    // 4 KB
  __shared__ _Float16 sH[64 * 128];    // 16 KB (h state, f16)
  const _Float16* gW = Whh16 + (size_t)n * 384 * 128;
  const float* bih = bih_all + n * 384;
  const float* bhh = bhh_all + n * 384;
  for (int s = tid; s < 384 * 32; s += 512) sWih[s] = Wih16[(size_t)n * 384 * 32 + s];
  for (int s = tid; s < 64 * 128; s += 512) sH[s] = (_Float16)0.f;
  __syncthreads();

  for (int t = 0; t < 256; ++t) {
    // Stage X[:, n, :, t] -> 64x32 f16 (C=16 zero-padded to 32).
    for (int s = tid; s < 64 * 32; s += 512) {
      int bb = s >> 5, c = s & 31;
      sXt[s] = (c < 16) ? (_Float16)X[((size_t)(bb * 16 + n) * 16 + c) * 256 + t]
                        : (_Float16)0.f;
    }
    __syncthreads();

    float hnew[2][8];
    int task[2] = {w, w + 16};
#pragma unroll 1
    for (int q = 0; q < 2; ++q) {
      int tm = task[q] & 3;        // batch tile (rows 16*tm..)
      int td = task[q] >> 2;       // hidden-col tile (cols 16*td..)
      const _Float16* Am = sXt + tm * 16 * 32;
      const _Float16* Ah = sH + tm * 16 * 128;
      v8f zero = {};
      v8f ar = wmma_acc(Am, 32, sWih + (td * 16) * 32, 32, 1, zero);
      ar = wmma_acc(Ah, 128, gW + (size_t)(td * 16) * 128, 128, 4, ar);
      v8f az = wmma_acc(Am, 32, sWih + ((td + 8) * 16) * 32, 32, 1, zero);
      az = wmma_acc(Ah, 128, gW + (size_t)((td + 8) * 16) * 128, 128, 4, az);
      v8f axn = wmma_acc(Am, 32, sWih + ((td + 16) * 16) * 32, 32, 1, zero);
      v8f ahn = wmma_acc(Ah, 128, gW + (size_t)((td + 16) * 16) * 128, 128, 4, zero);
      int lane = tid & 31, nl = lane & 15, half = lane >> 4;
      int d = td * 16 + nl;
#pragma unroll
      for (int r = 0; r < 8; ++r) {
        int m = tm * 16 + r + 8 * half;
        float hold = (float)sH[m * 128 + d];
        float rr = dsigmoid(ar[r] + bih[d] + bhh[d]);
        float zz = dsigmoid(az[r] + bih[128 + d] + bhh[128 + d]);
        float nn = tanhf(axn[r] + bih[256 + d] + rr * (ahn[r] + bhh[256 + d]));
        hnew[q][r] = (1.f - zz) * nn + zz * hold;
      }
    }
    __syncthreads();
#pragma unroll
    for (int q = 0; q < 2; ++q) {
      int tm = task[q] & 3, td = task[q] >> 2;
      int lane = tid & 31, nl = lane & 15, half = lane >> 4;
      int d = td * 16 + nl;
#pragma unroll
      for (int r = 0; r < 8; ++r)
        sH[(tm * 16 + r + 8 * half) * 128 + d] = (_Float16)hnew[q][r];
    }
    __syncthreads();

    // LayerNorm per row + store Hc[:,:,:,0:128]; 8 threads per batch row.
    {
      int row = tid >> 3, k8 = tid & 7;
      float s = 0.f, ss = 0.f;
      float vals[16];
#pragma unroll
      for (int i = 0; i < 16; ++i) {
        float hv = (float)sH[row * 128 + k8 * 16 + i];
        vals[i] = hv;
        s += hv;
        ss += hv * hv;
      }
      for (int m2 = 1; m2 < 8; m2 <<= 1) {
        s += __shfl_xor(s, m2);
        ss += __shfl_xor(ss, m2);
      }
      float mean = s * (1.f / 128.f);
      float var = ss * (1.f / 128.f) - mean * mean;
      float rstd = rsqrtf(var + 1e-5f);
      _Float16* outr = Hc + ((size_t)(row * 16 + n) * 256 + t) * 160;
#pragma unroll
      for (int i = 0; i < 16; ++i) {
        int d2 = k8 * 16 + i;
        outr[d2] = (_Float16)((vals[i] - mean) * rstd * lng[d2] + lnb[d2]);
      }
    }
    // u-features for Hc[:,:,:,128:160] (8 real, rest zero-pad).
    float tl = (float)t * (1.f / 255.f);
    for (int s2 = tid; s2 < 64 * 32; s2 += 512) {
      int bb = s2 >> 5, dp = s2 & 31;
      float uv = (dp < 8) ? tanhf(tl * tW[dp] + tb[dp]) : 0.f;
      Hc[((size_t)(bb * 16 + n) * 256 + t) * 160 + 128 + dp] = (_Float16)uv;
    }
    __syncthreads();
  }
}

// ---- Z = leaky(Hc@Wself^T + (A.Hc)@Wn^T); K = Z@Wk^T; V = Z@Wv^T -----------
// One block per (b, region i, 16-step t-block); 128 threads = 4 waves.
// Own-region tile staged via async global->LDS DMA, overlapped with the VALU
// graph-mix over the 16 neighbor tiles.
__global__ __launch_bounds__(128) void k_zkv(
    const _Float16* __restrict Hc, const float* __restrict A,
    const _Float16* __restrict Wself16, const _Float16* __restrict Wn16,
    const _Float16* __restrict Wk16, const _Float16* __restrict Wv16,
    _Float16* __restrict Kd, _Float16* __restrict Vd, float* __restrict h0) {
  int blk = blockIdx.x;
  int tb = blk & 15;
  int bn = blk >> 4;
  int i = bn & 15, b = bn >> 4;
  int tid = threadIdx.x;
  int w = tid >> 5;
  __shared__ _Float16 sHc[16 * 160];
  __shared__ _Float16 sHm[16 * 160];
  __shared__ _Float16 sZ[16 * 128];
  __shared__ float sAr[16];
  if (tid < 16) sAr[tid] = A[(b * 16 + i) * 16 + tid];
  int t0 = tb * 16;
  // Async-stage own tile: 16 rows x 160 f16 = 640 b64 transfers.
  const _Float16* own = Hc + ((size_t)(b * 16 + i) * 256 + t0) * 160;
#pragma unroll 1
  for (int s = tid; s < 640; s += 128) {
    int r = s / 40, q8 = s % 40;  // 40 x 8B per 160-f16 row
    async_ld_b64(&sHc[r * 160 + q8 * 4], own + (size_t)r * 160 + q8 * 4);
  }
  // Graph-mixed rows while the DMA is in flight: Hm = sum_j A[i,j] * Hc[b,j].
  for (int s = tid; s < 16 * 160; s += 128) {
    int r = s / 160, d = s % 160;
    float acc = 0.f;
#pragma unroll
    for (int j = 0; j < 16; ++j)
      acc += sAr[j] * (float)Hc[((size_t)(b * 16 + j) * 256 + t0 + r) * 160 + d];
    sHm[s] = (_Float16)acc;
  }
  async_wait0();
  __syncthreads();
  int lane = tid & 31, nl = lane & 15, half = lane >> 4;
#pragma unroll 1
  for (int q = 0; q < 2; ++q) {
    int nt = w + 4 * q;  // output d-tile 0..7
    v8f zero = {};
    v8f acc = wmma_acc(sHc, 160, Wself16 + nt * 16 * 160, 160, 5, zero);
    acc = wmma_acc(sHm, 160, Wn16 + nt * 16 * 160, 160, 5, acc);
#pragma unroll
    for (int r = 0; r < 8; ++r) {
      float e = acc[r];
      float z = e >= 0.f ? e : 0.1f * e;  // leaky relu
      int m = r + 8 * half;
      sZ[m * 128 + nt * 16 + nl] = (_Float16)z;
      if (tb == 15 && m == 15) h0[(b * 16 + i) * 128 + nt * 16 + nl] = z;
    }
  }
  __syncthreads();
#pragma unroll 1
  for (int q = 0; q < 2; ++q) {
    int nt = w + 4 * q;
    v8f zero = {};
    v8f ak = wmma_acc(sZ, 128, Wk16 + nt * 16 * 128, 128, 4, zero);
    v8f av = wmma_acc(sZ, 128, Wv16 + nt * 16 * 128, 128, 4, zero);
#pragma unroll
    for (int r = 0; r < 8; ++r) {
      int m = r + 8 * half;
      size_t o = ((size_t)(b * 16 + i) * 256 + t0 + m) * 128 + nt * 16 + nl;
      Kd[o] = (_Float16)ak[r];
      Vd[o] = (_Float16)av[r];
    }
  }
}

// ---- Decoder: 32 steps of graph-mixed attention GRU (one block per b) ------
__global__ __launch_bounds__(256) void k_dec(
    const float* __restrict X, const float* __restrict A,
    const _Float16* __restrict Kd, const _Float16* __restrict Vd,
    const float* __restrict h0, const _Float16* __restrict cWih16,
    const _Float16* __restrict cWhh16, const float* __restrict cbih,
    const float* __restrict cbhh, const _Float16* __restrict readW16,
    const float* __restrict readb, const float* __restrict log_tau,
    float* __restrict out) {
  int b = blockIdx.x;
  int tid = threadIdx.x;
  int w = tid >> 5;  // 8 waves
  __shared__ float sA[256];
  __shared__ float sh[16 * 128];        // h master (f32)
  __shared__ _Float16 sh16[16 * 128];   // h mirror for WMMA
  __shared__ float sy[256], sy2[256], syt[256];
  __shared__ float ssc[16 * 256];       // scores / alpha
  __shared__ float sctx[16 * 128];
  __shared__ _Float16 sinp[16 * 160];   // [ctx(128) | y(16) | pad]
  __shared__ float redA[256], redB[256];

  sA[tid] = A[b * 256 + tid];
  sy[tid] = 0.f;
  for (int s = tid; s < 2048; s += 256) {
    float hv = h0[b * 2048 + s];
    sh[s] = hv;
    sh16[s] = (_Float16)hv;
  }
  float scale = __expf(log_tau[0]);
  __syncthreads();

  for (int step = 0; step < 32; ++step) {
    // y <- y + 0.3 * A @ y
    {
      int i = tid >> 4, c = tid & 15;
      float acc = 0.f;
#pragma unroll
      for (int j = 0; j < 16; ++j) acc += sA[i * 16 + j] * sy[j * 16 + c];
      sy2[tid] = sy[tid] + 0.3f * acc;
    }
    __syncthreads();
    // scores[j,t] = (K[b,j,t,:] . h[j]) * scale
    for (int s = tid; s < 4096; s += 256) {
      int j = s >> 8, t = s & 255;
      const _Float16* kr = Kd + ((size_t)(b * 16 + j) * 256 + t) * 128;
      __builtin_prefetch(kr + 16 * 128, 0, 0);  // global_prefetch_b8
      float acc = 0.f;
#pragma unroll 4
      for (int d = 0; d < 128; ++d) acc += (float)kr[d] * sh[j * 128 + d];
      ssc[s] = acc * scale;
    }
    __syncthreads();
    // softmax over t per region (16 threads/region).
    {
      int g = tid >> 4, l = tid & 15;
      float m = -1e30f;
#pragma unroll
      for (int k = 0; k < 16; ++k) m = fmaxf(m, ssc[g * 256 + l * 16 + k]);
      redA[tid] = m;
      __syncthreads();
      float M = -1e30f;
#pragma unroll
      for (int k = 0; k < 16; ++k) M = fmaxf(M, redA[g * 16 + k]);
      float ps = 0.f;
#pragma unroll
      for (int k = 0; k < 16; ++k) {
        float e = __expf(ssc[g * 256 + l * 16 + k] - M);
        ssc[g * 256 + l * 16 + k] = e;
        ps += e;
      }
      redB[tid] = ps;
      __syncthreads();
      float S_ = 0.f;
#pragma unroll
      for (int k = 0; k < 16; ++k) S_ += redB[g * 16 + k];
      float inv = 1.f / S_;
#pragma unroll
      for (int k = 0; k < 16; ++k) ssc[g * 256 + l * 16 + k] *= inv;
    }
    __syncthreads();
    // ctx[j,:] = alpha[j,:] @ V[b,j,:,:]
    for (int s = tid; s < 2048; s += 256) {
      int j = s >> 7, d = s & 127;
      const _Float16* vb = Vd + ((size_t)(b * 16 + j) * 256) * 128 + d;
      float acc = 0.f;
      for (int t = 0; t < 256; ++t) acc += ssc[j * 256 + t] * (float)vb[(size_t)t * 128];
      sctx[s] = acc;
    }
    __syncthreads();
    // inp = concat(ctx, y_mixed) padded to 160.
    for (int s = tid; s < 2560; s += 256) {
      int j = s / 160, d = s % 160;
      float v2 = (d < 128) ? sctx[j * 128 + d]
                           : ((d < 144) ? sy2[j * 16 + (d - 128)] : 0.f);
      sinp[s] = (_Float16)v2;
    }
    __syncthreads();
    // GRU cell: gx = inp@cWih^T (K=160), gh = h@cWhh^T (K=128); wave w -> d-tile w.
    float hnew[8];
    {
      int td = w;
      v8f zero = {};
      v8f ar = wmma_acc(sinp, 160, cWih16 + (td * 16) * 160, 160, 5, zero);
      ar = wmma_acc(sh16, 128, cWhh16 + (td * 16) * 128, 128, 4, ar);
      v8f az = wmma_acc(sinp, 160, cWih16 + ((td + 8) * 16) * 160, 160, 5, zero);
      az = wmma_acc(sh16, 128, cWhh16 + ((td + 8) * 16) * 128, 128, 4, az);
      v8f axn = wmma_acc(sinp, 160, cWih16 + ((td + 16) * 16) * 160, 160, 5, zero);
      v8f ahn = wmma_acc(sh16, 128, cWhh16 + ((td + 16) * 16) * 128, 128, 4, zero);
      int lane = tid & 31, nl = lane & 15, half = lane >> 4;
      int d = td * 16 + nl;
#pragma unroll
      for (int r = 0; r < 8; ++r) {
        int m = r + 8 * half;  // region
        float hold = sh[m * 128 + d];
        float rr = dsigmoid(ar[r] + cbih[d] + cbhh[d]);
        float zz = dsigmoid(az[r] + cbih[128 + d] + cbhh[128 + d]);
        float nn = tanhf(axn[r] + cbih[256 + d] + rr * (ahn[r] + cbhh[256 + d]));
        hnew[r] = (1.f - zz) * nn + zz * hold;
      }
    }
    __syncthreads();
    {
      int lane = tid & 31, nl = lane & 15, half = lane >> 4;
      int d = w * 16 + nl;
#pragma unroll
      for (int r = 0; r < 8; ++r) {
        int m = r + 8 * half;
        sh[m * 128 + d] = hnew[r];
        sh16[m * 128 + d] = (_Float16)hnew[r];
      }
    }
    __syncthreads();
    // Readout: yt = h@read_W^T + read_b + y_mixed (single 16x16 tile, wave 0).
    if (w == 0) {
      v8f zero = {};
      v8f ay = wmma_acc(sh16, 128, readW16, 128, 4, zero);
      int lane = tid & 31, nl = lane & 15, half = lane >> 4;
#pragma unroll
      for (int r = 0; r < 8; ++r) {
        int m = r + 8 * half;
        syt[m * 16 + nl] = ay[r] + readb[nl] + sy2[m * 16 + nl];
      }
    }
    __syncthreads();
    // Commit: carry y, write Y_hat = X[...,-1] + yt.
    {
      int i = tid >> 4, c = tid & 15;
      float v2 = syt[tid];
      sy[tid] = v2;
      float xl = X[((size_t)(b * 16 + i) * 16 + c) * 256 + 255];
      float yh = xl + v2;
      out[((size_t)(b * 16 + i) * 16 + c) * 32 + step] = yh;
      if (step == 0) out[524288 + (size_t)(b * 16 + i) * 16 + c] = yh;
    }
    __syncthreads();
  }
}

// ---- launcher --------------------------------------------------------------

extern "C" void kernel_launch(void* const* d_in, const int* in_sizes, int n_in,
                              void* d_out, int out_size, void* d_ws, size_t ws_size,
                              hipStream_t stream) {
  const float* X        = (const float*)d_in[0];
  const int*   phases   = (const int*)d_in[1];
  const float* gru_Wih  = (const float*)d_in[2];
  const float* gru_Whh  = (const float*)d_in[3];
  const float* gru_bih  = (const float*)d_in[4];
  const float* gru_bhh  = (const float*)d_in[5];
  const float* ln_g     = (const float*)d_in[6];
  const float* ln_b     = (const float*)d_in[7];
  const float* tW       = (const float*)d_in[8];
  const float* tb       = (const float*)d_in[9];
  const float* S        = (const float*)d_in[10];
  const float* G        = (const float*)d_in[11];
  const float* W_self   = (const float*)d_in[12];
  const float* W_neigh  = (const float*)d_in[13];
  const float* cell_Wih = (const float*)d_in[14];
  const float* cell_Whh = (const float*)d_in[15];
  const float* cell_bih = (const float*)d_in[16];
  const float* cell_bhh = (const float*)d_in[17];
  const float* read_W   = (const float*)d_in[18];
  const float* read_b   = (const float*)d_in[19];
  const float* Wk       = (const float*)d_in[20];
  const float* Wv       = (const float*)d_in[21];
  const float* log_tau  = (const float*)d_in[22];

  char* ws = (char*)d_ws;
  float*    wA       = (float*)(ws + 0);
  _Float16* Wn16     = (_Float16*)(ws + 65536);
  _Float16* Wself16  = (_Float16*)(ws + 106496);
  _Float16* Wih16    = (_Float16*)(ws + 147456);
  _Float16* Whh16    = (_Float16*)(ws + 540672);
  _Float16* Wk16     = (_Float16*)(ws + 2113536);
  _Float16* Wv16     = (_Float16*)(ws + 2146304);
  _Float16* cWih16   = (_Float16*)(ws + 2179072);
  _Float16* cWhh16   = (_Float16*)(ws + 2301952);
  _Float16* readW16  = (_Float16*)(ws + 2400256);
  float*    h0w      = (float*)(ws + 2404352);
  _Float16* Hc       = (_Float16*)(ws + 2928640);
  _Float16* Kd       = (_Float16*)(ws + 86814720);
  _Float16* Vd       = (_Float16*)(ws + 153923584);

  auto cdiv = [](int a, int b) { return (a + b - 1) / b; };

  k_graphs<<<64, 256, 0, stream>>>(S, G, phases, wA);
  k_spectral<<<1, 256, 0, stream>>>(W_neigh, Wn16);
  k_padcvt<<<cdiv(128 * 160, 256), 256, 0, stream>>>(W_self, Wself16, 128, 136, 160);
  k_padcvt<<<cdiv(16 * 384 * 32, 256), 256, 0, stream>>>(gru_Wih, Wih16, 16 * 384, 16, 32);
  k_padcvt<<<cdiv(16 * 384 * 128, 256), 256, 0, stream>>>(gru_Whh, Whh16, 16 * 384, 128, 128);
  k_padcvt<<<cdiv(128 * 128, 256), 256, 0, stream>>>(Wk, Wk16, 128, 128, 128);
  k_padcvt<<<cdiv(128 * 128, 256), 256, 0, stream>>>(Wv, Wv16, 128, 128, 128);
  k_padcvt<<<cdiv(384 * 160, 256), 256, 0, stream>>>(cell_Wih, cWih16, 384, 144, 160);
  k_padcvt<<<cdiv(384 * 128, 256), 256, 0, stream>>>(cell_Whh, cWhh16, 384, 128, 128);
  k_padcvt<<<cdiv(16 * 128, 256), 256, 0, stream>>>(read_W, readW16, 16, 128, 128);

  k_gru<<<16, 512, 0, stream>>>(X, Wih16, Whh16, gru_bih, gru_bhh, ln_g, ln_b,
                                tW, tb, Hc);
  k_zkv<<<16384, 128, 0, stream>>>(Hc, wA, Wself16, Wn16, Wk16, Wv16, Kd, Vd, h0w);
  k_dec<<<64, 256, 0, stream>>>(X, wA, Kd, Vd, h0w, cWih16, cWhh16, cell_bih,
                                cell_bhh, readW16, read_b, log_tau, (float*)d_out);
}
